// S6LayerStateful_18657337934049
// MI455X (gfx1250) — compile-verified, matrix operand order checked by reading, not run on hardware
//
#include <hip/hip_runtime.h>

// ---------------------------------------------------------------------------
// Mamba S6 layer for gfx1250 (MI455X): bf16 WMMA GEMMs (double-buffered,
// 32x64 wave tiles) + LDS-staged register-state selective scan.
// ---------------------------------------------------------------------------

#define BATCH   4
#define SEQ     2048
#define D_MODEL 1024
#define D_INNER 2048
#define D_STATE 16
#define D_CONV  4
#define DT_RANK 64
#define NPROJ   (DT_RANK + 2 * D_STATE)   // 96
#define ROWS    (BATCH * SEQ)             // 8192

typedef __attribute__((ext_vector_type(16))) __bf16 v16bf;
typedef __attribute__((ext_vector_type(8)))  __bf16 v8bf;
typedef __attribute__((ext_vector_type(8)))  float  v8f;

__device__ __forceinline__ __bf16 f32_to_bf16(float f) {
    unsigned u = __float_as_uint(f);
    unsigned r = (u + 0x7FFFu + ((u >> 16) & 1u)) >> 16;   // round-to-nearest-even
    unsigned short s = (unsigned short)r;
    return __builtin_bit_cast(__bf16, s);
}

__device__ __forceinline__ float silu(float x) {
    return x / (1.0f + __expf(-x));
}

// 16-bit WMMA operand tile: lane m (0..15) holds K {0..7,16..23},
// lane m+16 holds K {8..15,24..31} -> two 16B loads per tile.
__device__ __forceinline__ v16bf load_tile16(const __bf16* __restrict__ p) {
    v8bf lo = *(const v8bf*)p;
    v8bf hi = *(const v8bf*)(p + 16);
    return __builtin_shufflevector(lo, hi, 0,1,2,3,4,5,6,7,8,9,10,11,12,13,14,15);
}

// ---------------------------------------------------------------------------
// f32 -> bf16 conversion kernels
// ---------------------------------------------------------------------------
__global__ __launch_bounds__(256) void cvt_f32_to_bf16(const float* __restrict__ src,
                                                       __bf16* __restrict__ dst, int n) {
    int i = blockIdx.x * blockDim.x + threadIdx.x;
    if (i < n) dst[i] = f32_to_bf16(src[i]);
}

__global__ __launch_bounds__(256) void cvt_f32_to_bf16_strided(const float* __restrict__ src,
                                                               __bf16* __restrict__ dst,
                                                               int rows, int cols, int srcStride) {
    int i = blockIdx.x * blockDim.x + threadIdx.x;
    if (i >= rows * cols) return;
    int r = i / cols, c = i - r * cols;
    dst[i] = f32_to_bf16(src[(size_t)r * srcStride + c]);
}

// ---------------------------------------------------------------------------
// bf16 WMMA GEMM:  C[M,N] (f32) = A[M,K] * B[N,K]^T     (A,B row-major bf16)
// One wave computes a (16*MT) x (16*NT) tile. K stepped by 32 per v_wmma,
// software-pipelined two K-steps deep so loads overlap WMMAs.
// Requires K % 64 == 0.
// ---------------------------------------------------------------------------
template <int MT, int NT>
__global__ __launch_bounds__(256) void gemm_bf16_wmma(const __bf16* __restrict__ A,
                                                      const __bf16* __restrict__ B,
                                                      float* __restrict__ C,
                                                      int M, int N, int K) {
    const int lane  = threadIdx.x & 31;
    const int wave  = (blockIdx.x * blockDim.x + threadIdx.x) >> 5;
    const int tilesN = N / (16 * NT);
    const int tilesM = M / (16 * MT);
    if (wave >= tilesM * tilesN) return;

    const int tm = wave / tilesN;
    const int tn = wave - tm * tilesN;
    const int lm = lane & 15;
    const int kh = lane >> 4;            // 0: K 0..7/16..23   1: K 8..15/24..31

    const __bf16* pa[MT];
#pragma unroll
    for (int mt = 0; mt < MT; ++mt)
        pa[mt] = A + (size_t)(((tm * MT + mt) << 4) + lm) * K + (kh << 3);
    const __bf16* pb[NT];
#pragma unroll
    for (int t = 0; t < NT; ++t)
        pb[t] = B + (size_t)(((tn * NT + t) << 4) + lm) * K + (kh << 3);

    v8f acc[MT][NT];
    const v8f vzero = {0.f, 0.f, 0.f, 0.f, 0.f, 0.f, 0.f, 0.f};
#pragma unroll
    for (int mt = 0; mt < MT; ++mt)
#pragma unroll
        for (int t = 0; t < NT; ++t) acc[mt][t] = vzero;

    v16bf a0[MT], b0[NT], a1[MT], b1[NT];
#pragma unroll
    for (int mt = 0; mt < MT; ++mt) a0[mt] = load_tile16(pa[mt]);
#pragma unroll
    for (int t = 0; t < NT; ++t)    b0[t]  = load_tile16(pb[t]);

    for (int k0 = 0; k0 + 64 <= K; k0 += 64) {
        // prefetch K-step k0+32 while computing on k0
#pragma unroll
        for (int mt = 0; mt < MT; ++mt) a1[mt] = load_tile16(pa[mt] + k0 + 32);
#pragma unroll
        for (int t = 0; t < NT; ++t)    b1[t]  = load_tile16(pb[t] + k0 + 32);
#pragma unroll
        for (int mt = 0; mt < MT; ++mt)
#pragma unroll
            for (int t = 0; t < NT; ++t)
                acc[mt][t] = __builtin_amdgcn_wmma_f32_16x16x32_bf16(
                    false, a0[mt], false, b0[t], (short)0, acc[mt][t], false, false);

        if (k0 + 64 < K) {   // prefetch K-step k0+64 while computing on k0+32
#pragma unroll
            for (int mt = 0; mt < MT; ++mt) a0[mt] = load_tile16(pa[mt] + k0 + 64);
#pragma unroll
            for (int t = 0; t < NT; ++t)    b0[t]  = load_tile16(pb[t] + k0 + 64);
        }
#pragma unroll
        for (int mt = 0; mt < MT; ++mt)
#pragma unroll
            for (int t = 0; t < NT; ++t)
                acc[mt][t] = __builtin_amdgcn_wmma_f32_16x16x32_bf16(
                    false, a1[mt], false, b1[t], (short)0, acc[mt][t], false, false);
    }

    // C/D layout: VGPR j -> row = tile*16 + j + 8*kh, col = lane&15 within tile
#pragma unroll
    for (int mt = 0; mt < MT; ++mt) {
#pragma unroll
        for (int t = 0; t < NT; ++t) {
            const int ncol = ((tn * NT + t) << 4) + lm;
#pragma unroll
            for (int j = 0; j < 8; ++j) {
                const int r = ((tm * MT + mt) << 4) + j + (kh << 3);
                C[(size_t)r * N + ncol] = acc[mt][t][j];
            }
        }
    }
}

// ---------------------------------------------------------------------------
// Depthwise causal conv1d (k=4) + SiLU. Reads x_p = xz[..., 0:D_INNER].
// Writes f32 (for the scan) and bf16 (for the x_proj GEMM).
// ---------------------------------------------------------------------------
__global__ __launch_bounds__(256) void conv_silu_kernel(const float* __restrict__ xz,
                                                        const float* __restrict__ conv_w,
                                                        const float* __restrict__ conv_b,
                                                        float* __restrict__ xc,
                                                        __bf16* __restrict__ xcbf) {
    int idx = blockIdx.x * blockDim.x + threadIdx.x;    // over B*L*D_INNER
    if (idx >= BATCH * SEQ * D_INNER) return;
    int d = idx & (D_INNER - 1);
    int l = (idx >> 11) & (SEQ - 1);
    int b = idx >> 22;

    float acc = conv_b[d];
#pragma unroll
    for (int j = 0; j < D_CONV; ++j) {
        int ls = l - (D_CONV - 1) + j;
        if (ls >= 0)
            acc += conv_w[d * D_CONV + j] *
                   xz[((size_t)b * SEQ + ls) * (2 * D_INNER) + d];
    }
    float s = silu(acc);
    xc[idx]   = s;
    xcbf[idx] = f32_to_bf16(s);
}

// ---------------------------------------------------------------------------
// Selective scan: one lane per (batch, channel); 16 states in registers.
// B/C vectors staged through LDS in 8-step chunks (shared by all 256 channels
// of the block). Folds softplus(dt) and the silu(z) gate; emits bf16 Y.
// ---------------------------------------------------------------------------
#define SCAN_CHUNK 8

__global__ __launch_bounds__(256) void scan_kernel(const float* __restrict__ xz,
                                                   const float* __restrict__ xc,
                                                   const float* __restrict__ dtlin,
                                                   const float* __restrict__ xdbl,
                                                   const float* __restrict__ dt_bias,
                                                   const float* __restrict__ A_log,
                                                   const float* __restrict__ Dvec,
                                                   __bf16* __restrict__ Ybf) {
    __shared__ float sBC[SCAN_CHUNK * 2 * D_STATE];   // 8 steps x 32 floats

    const int tid = threadIdx.x;
    const int b   = blockIdx.x >> 3;                  // 4 batches x 8 channel groups
    const int d   = ((blockIdx.x & 7) << 8) + tid;    // 256 channels per block

    const float bias = dt_bias[d];
    const float Dd   = Dvec[d];

    float Av[D_STATE], h[D_STATE];
#pragma unroll
    for (int n = 0; n < D_STATE; ++n) {
        Av[n] = -__expf(A_log[d * D_STATE + n]);
        h[n]  = 0.0f;
    }

    const size_t rowBase = (size_t)b * SEQ;
    for (int l0 = 0; l0 < SEQ; l0 += SCAN_CHUNK) {
        __syncthreads();
        {   // cooperative load: 8 rows x 32 B/C elements = 256 floats
            const int r = tid >> 5, e = tid & 31;
            sBC[tid] = xdbl[(rowBase + l0 + r) * NPROJ + DT_RANK + e];
        }
        __syncthreads();

#pragma unroll
        for (int li = 0; li < SCAN_CHUNK; ++li) {
            const size_t row = rowBase + l0 + li;
            const float xv  = xc[row * D_INNER + d];
            const float dtl = dtlin[row * D_INNER + d] + bias;
            const float dt  = (dtl > 20.0f) ? dtl : log1pf(__expf(dtl));
            const float* bc = &sBC[li * 2 * D_STATE]; // B[0..15], C[16..31]

            float y = 0.0f;
#pragma unroll
            for (int n = 0; n < D_STATE; ++n) {
                float dtA = dt * Av[n];
                dtA = fminf(fmaxf(dtA, -20.0f), 0.0f);
                float Ab = __expf(dtA);
                float hn = Ab * h[n] + dt * bc[n] * xv;
                hn = fminf(fmaxf(hn, -100.0f), 100.0f);
                h[n] = hn;
                y += hn * bc[D_STATE + n];
            }
            y += Dd * xv;

            const float z = xz[row * (2 * D_INNER) + D_INNER + d];
            Ybf[row * D_INNER + d] = f32_to_bf16(y * silu(z));
        }
    }
}

// ---------------------------------------------------------------------------
// Host launcher
// ---------------------------------------------------------------------------
static inline int gemm_blocks(int M, int N, int MT, int NT) {
    long waves   = (long)(M / (16 * MT)) * (N / (16 * NT));
    long threads = waves * 32;
    return (int)((threads + 255) / 256);
}

extern "C" void kernel_launch(void* const* d_in, const int* in_sizes, int n_in,
                              void* d_out, int out_size, void* d_ws, size_t ws_size,
                              hipStream_t stream) {
    const float* x         = (const float*)d_in[0];  // [B,L,D_MODEL]
    const float* in_proj_w = (const float*)d_in[1];  // [2*D_INNER, D_MODEL]
    const float* conv_w    = (const float*)d_in[2];  // [D_INNER,1,D_CONV]
    const float* conv_b    = (const float*)d_in[3];  // [D_INNER]
    const float* x_proj_w  = (const float*)d_in[4];  // [NPROJ, D_INNER]
    const float* dt_proj_w = (const float*)d_in[5];  // [D_INNER, DT_RANK]
    const float* dt_proj_b = (const float*)d_in[6];  // [D_INNER]
    const float* A_log     = (const float*)d_in[7];  // [D_INNER, D_STATE]
    const float* Dvec      = (const float*)d_in[8];  // [D_INNER]
    const float* out_proj_w= (const float*)d_in[9];  // [D_MODEL, D_INNER]
    float* out = (float*)d_out;                      // [B,L,D_MODEL]

    // --- carve workspace ---
    char* ws = (char*)d_ws;
    size_t off = 0;
    auto carve = [&](size_t bytes) -> void* {
        void* p = ws + off;
        off += (bytes + 255) & ~(size_t)255;
        return p;
    };
    __bf16* Xbf    = (__bf16*)carve((size_t)ROWS * D_MODEL * 2);
    __bf16* Winbf  = (__bf16*)carve((size_t)2 * D_INNER * D_MODEL * 2);
    float*  xz     = (float*) carve((size_t)ROWS * 2 * D_INNER * 4);
    float*  xcf    = (float*) carve((size_t)ROWS * D_INNER * 4);
    __bf16* xcbf   = (__bf16*)carve((size_t)ROWS * D_INNER * 2);
    __bf16* Wxpbf  = (__bf16*)carve((size_t)NPROJ * D_INNER * 2);
    float*  xdbl   = (float*) carve((size_t)ROWS * NPROJ * 4);
    __bf16* DTbf   = (__bf16*)carve((size_t)ROWS * DT_RANK * 2);
    __bf16* Wdtbf  = (__bf16*)carve((size_t)D_INNER * DT_RANK * 2);
    float*  dtlin  = (float*) carve((size_t)ROWS * D_INNER * 4);
    __bf16* Ybf    = (__bf16*)carve((size_t)ROWS * D_INNER * 2);
    __bf16* Woutbf = (__bf16*)carve((size_t)D_MODEL * D_INNER * 2);
    (void)ws_size;

    const int TB = 256;
    auto blk = [](int n) { return (n + 255) / 256; };

    // 1) bf16 conversions of activations + weights
    cvt_f32_to_bf16<<<blk(ROWS * D_MODEL), TB, 0, stream>>>(x, Xbf, ROWS * D_MODEL);
    cvt_f32_to_bf16<<<blk(2 * D_INNER * D_MODEL), TB, 0, stream>>>(in_proj_w, Winbf, 2 * D_INNER * D_MODEL);
    cvt_f32_to_bf16<<<blk(NPROJ * D_INNER), TB, 0, stream>>>(x_proj_w, Wxpbf, NPROJ * D_INNER);
    cvt_f32_to_bf16<<<blk(D_INNER * DT_RANK), TB, 0, stream>>>(dt_proj_w, Wdtbf, D_INNER * DT_RANK);
    cvt_f32_to_bf16<<<blk(D_MODEL * D_INNER), TB, 0, stream>>>(out_proj_w, Woutbf, D_MODEL * D_INNER);

    // 2) in_proj: xz = X @ Win^T      [8192,1024] x [4096,1024]^T
    gemm_bf16_wmma<2,4><<<gemm_blocks(ROWS, 2 * D_INNER, 2, 4), TB, 0, stream>>>(
        Xbf, Winbf, xz, ROWS, 2 * D_INNER, D_MODEL);

    // 3) causal depthwise conv + SiLU
    conv_silu_kernel<<<blk(BATCH * SEQ * D_INNER), TB, 0, stream>>>(
        xz, conv_w, conv_b, xcf, xcbf);

    // 4) x_proj: x_dbl = xc @ Wxp^T   [8192,2048] x [96,2048]^T
    gemm_bf16_wmma<2,2><<<gemm_blocks(ROWS, NPROJ, 2, 2), TB, 0, stream>>>(
        xcbf, Wxpbf, xdbl, ROWS, NPROJ, D_INNER);

    // 5) dt slice -> bf16, then dt_proj: dtlin = dt @ Wdt^T  [8192,64] x [2048,64]^T
    cvt_f32_to_bf16_strided<<<blk(ROWS * DT_RANK), TB, 0, stream>>>(
        xdbl, DTbf, ROWS, DT_RANK, NPROJ);
    gemm_bf16_wmma<2,4><<<gemm_blocks(ROWS, D_INNER, 2, 4), TB, 0, stream>>>(
        DTbf, Wdtbf, dtlin, ROWS, D_INNER, DT_RANK);

    // 6) selective scan + gate  ->  Ybf   (32 blocks: 4 batches x 8 channel groups)
    scan_kernel<<<BATCH * (D_INNER / 256), TB, 0, stream>>>(
        xz, xcf, dtlin, xdbl, dt_proj_b, A_log, Dvec, Ybf);

    // 7) out_proj: out = Y @ Wout^T   [8192,2048] x [1024,2048]^T
    gemm_bf16_wmma<2,4><<<gemm_blocks(ROWS, D_MODEL, 2, 4), TB, 0, stream>>>(
        Ybf, Woutbf, out, ROWS, D_MODEL, D_INNER);

    (void)in_sizes; (void)n_in; (void)out_size;
}
